// MKGN_21852793602343
// MI455X (gfx1250) — matrix operand order repaired and checked by reading, not compiled, not run on hardware
//
#include <hip/hip_runtime.h>
#include <hip/hip_bf16.h>

// ============================================================================
// MGKN forward for MI455X (gfx1250, wave32, WMMA).
//
// Strategy (roofline, 23.3 TB/s HBM):
//  * Per-edge kernel MLPs (~34 GFLOP, GEMM-shaped) -> v_wmma_f32_16x16x32_f16
//    with f32 accumulate, fused bias+ReLU, weights pre-transposed to [N,K] f16
//    so A and B fragments are contiguous b128 loads. Tile count per wave is a
//    TEMPLATE parameter -> zero branches in the K loop, EXEC all-1s, no spills.
//  * We matrices stored f16 (386 MB written once, read 2x) -> traffic-bound,
//    ~50us of HBM time. Wt weight matrices (<256KB each) live in the 192MB L2.
//  * Per-edge 32x32 matvec conv: wave-per-edge, __shfl broadcast of source
//    vector, coalesced f16 We column loads, global_atomic_add_f32 scatter.
//
// Input index map (insertion-order flatten of setup_inputs(), params = 43
// leaves in dict/list insertion order):
//  0 x | 1 ea_down | 2 ea_mid | 3 ea_up
//  4 fc_in.W  5 fc_in.b
//  6..9   down0 (W1,b1,W2,b2)   10..13 down1
//  14..19 mid0 (W1,b1,W2,b2,W3,b3)  20..25 mid1  26..31 mid2
//  32..34 mid_root[0..2]
//  35..38 up0  39..42 up1
//  43 fc_out1.W 44 fc_out1.b 45 fc_out2.W 46 fc_out2.b
//  47 edge_index_down [2,40960] | 48 edge_index_mid [2,106496] | 49 edge_index_up [2,40960]
//  50..53 range_*/points (static; unused)
// ============================================================================

typedef __attribute__((ext_vector_type(16))) _Float16 v16h;
typedef __attribute__((ext_vector_type(8)))  _Float16 v8h;
typedef __attribute__((ext_vector_type(8)))  float    v8f;

#define CH 32   // node feature width

// ---------------- fc_in: h[n,o] = x[n]*W[o]+b[o] ----------------------------
__global__ void k_fc_in(const float* __restrict__ x, const float* __restrict__ W,
                        const float* __restrict__ b, float* __restrict__ h, int n)
{
    int i = blockIdx.x * blockDim.x + threadIdx.x;
    if (i >= n * CH) return;
    int node = i >> 5, o = i & 31;
    h[i] = x[node] * W[o] + b[o];
}

// ---------------- weight prep: f32 W[K,N] -> f16 Wt[N,K] --------------------
__global__ void k_w2h_t(const float* __restrict__ W, _Float16* __restrict__ Wt,
                        int K, int N)
{
    int i = blockIdx.x * blockDim.x + threadIdx.x;
    if (i >= K * N) return;
    int k = i / N, n = i - k * N;
    Wt[(size_t)n * K + k] = (_Float16)W[i];
}

// ---------------- MLP layer 1 (K=6, tiny): f32 in -> ReLU -> f16 out --------
__global__ void k_mlp_l1(const float* __restrict__ attr, const float* __restrict__ W,
                         const float* __restrict__ b, _Float16* __restrict__ Y,
                         int E, int H)
{
    int i = blockIdx.x * blockDim.x + threadIdx.x;
    if (i >= E * H) return;
    int e = i / H, o = i - e * H;
    float a = b[o];
    const float* ap = attr + (size_t)e * 6;
#pragma unroll
    for (int j = 0; j < 6; ++j) a += ap[j] * W[j * H + o];
    Y[i] = (_Float16)fmaxf(a, 0.f);
}

// ---------------- WMMA GEMM: Y[M,N] = relu?(X[M,K]f16 @ Wt[N,K]f16 + b) -----
// M multiple of 16; K multiple of 32; (N/16) % NT == 0 (enforced by host).
// One wave owns a 16-row strip and NT consecutive 16-col tiles (A fragment
// reused NT times). NT is compile-time -> the K loop and epilogue contain NO
// branches, EXEC stays all-1s at every v_wmma (ISA 7.12), and the
// accumulators never cross a divergent region (no scratch spills).
//
// A fragment (16x32; contiguous k order used here, not the ISA table order): row = lane&15, half = lane>>4;
//   elements 0..7  -> K = k0 + half*16 + e       (contiguous -> v8h load)
//   elements 8..15 -> K = k0 + half*16 + e       (contiguous -> v8h load)
// B fragment (32x16, K-major via pre-transposed Wt): col = lane&15;
//   elements 0..15 -> K = k0 + half*16 + e       (contiguous -> 32B load)
// C/D: element r -> row = r + half*8, col = lane&15.
template<int NT, bool RELU>
__global__ void k_gemm_wmma(const _Float16* __restrict__ X, const _Float16* __restrict__ Wt,
                            const float* __restrict__ bias, _Float16* __restrict__ Y,
                            int M, int K, int N, int ngroups)
{
    int wid  = (blockIdx.x * blockDim.x + threadIdx.x) >> 5;
    int lane = threadIdx.x & 31;
    int mt = wid / ngroups;
    int ng = wid - mt * ngroups;
    if (mt * 16 >= M) return;   // wave-uniform exit only

    const int half = lane >> 4;
    const int l15  = lane & 15;
    const _Float16* arow  = X  + (size_t)(mt * 16 + l15) * K + half * 16;
    const _Float16* bbase = Wt + (size_t)(ng * NT * 16 + l15) * K + half * 16;

    v8f acc[NT];
#pragma unroll
    for (int t = 0; t < NT; ++t) acc[t] = v8f{};

    for (int k0 = 0; k0 < K; k0 += 32) {
        v8h alo = *(const v8h*)(arow + k0);        // K = k0 + half*16 .. +7
        v8h ahi = *(const v8h*)(arow + k0 + 8);    // K = k0+8+half*16 .. +7
        v16h a = __builtin_shufflevector(alo, ahi,
                  0, 1, 2, 3, 4, 5, 6, 7, 8, 9, 10, 11, 12, 13, 14, 15);
#pragma unroll
        for (int t = 0; t < NT; ++t) {
            v16h bfrag = *(const v16h*)(bbase + (size_t)t * 16 * K + k0);
            acc[t] = __builtin_amdgcn_wmma_f32_16x16x32_f16(
                false, a, false, bfrag, (short)0, acc[t], false, false);
        }
    }

#pragma unroll
    for (int t = 0; t < NT; ++t) {
        int col = (ng * NT + t) * 16 + l15;
        float bv = bias[col];
#pragma unroll
        for (int r = 0; r < 8; ++r) {
            int row = mt * 16 + r + half * 8;
            float v = acc[t][r] + bv;
            if (RELU) v = fmaxf(v, 0.f);
            Y[(size_t)row * N + col] = (_Float16)v;
        }
    }
}

// ---------------- scatter-degree count --------------------------------------
__global__ void k_count(const int* __restrict__ tgt, float* __restrict__ cnt, int E)
{
    int i = blockIdx.x * blockDim.x + threadIdx.x;
    if (i < E) atomicAdd(&cnt[tgt[i]], 1.f);
}

// ---------------- NNConv message + scatter-add ------------------------------
// wave per edge; lane = output channel. msg[o] = sum_i h[src][i] * We[e,i,o].
__global__ void k_conv(const float* __restrict__ h, const int* __restrict__ src,
                       const int* __restrict__ tgt, const _Float16* __restrict__ We,
                       float* __restrict__ s, int E)
{
    int wid  = (blockIdx.x * blockDim.x + threadIdx.x) >> 5;
    int lane = threadIdx.x & 31;
    if (wid >= E) return;
    float xv = h[(size_t)src[wid] * CH + lane];
    const _Float16* w = We + (size_t)wid * (CH * CH) + lane;
    float acc = 0.f;
#pragma unroll
    for (int i = 0; i < CH; ++i)
        acc += __shfl(xv, i, 32) * (float)w[i * CH];
    atomicAdd(&s[(size_t)tgt[wid] * CH + lane], acc);
}

// ---------------- down/up combine: h = relu(h + s/max(c,1)) over ALL nodes --
__global__ void k_combine_relu(float* __restrict__ h, const float* __restrict__ s,
                               const float* __restrict__ cnt, int total)
{
    int i = blockIdx.x * blockDim.x + threadIdx.x;
    if (i >= total) return;
    float c = fmaxf(cnt[i >> 5], 1.f);
    h[i] = fmaxf(h[i] + s[i] / c, 0.f);
}

// ---------------- mid combine: h[p0+k] = s/max(c,1) + h[p0+k] @ root --------
__global__ void k_combine_mid(float* __restrict__ h, const float* __restrict__ s,
                              const float* __restrict__ cnt, const float* __restrict__ root,
                              int p0, int nnodes)
{
    int wid  = (blockIdx.x * blockDim.x + threadIdx.x) >> 5;
    int lane = threadIdx.x & 31;
    if (wid >= nnodes) return;
    size_t base = (size_t)(p0 + wid) * CH;
    float xv  = h[base + lane];
    float acc = s[base + lane] / fmaxf(cnt[p0 + wid], 1.f);
#pragma unroll
    for (int i = 0; i < CH; ++i)
        acc += __shfl(xv, i, 32) * root[i * CH + lane];
    h[base + lane] = acc;   // overwrite, no ReLU (matches reference)
}

// ---------------- fused output head: relu(h@W1+b1)@W2+b2 --------------------
__global__ void k_fc_out(const float* __restrict__ h, const float* __restrict__ W1,
                         const float* __restrict__ b1, const float* __restrict__ W2,
                         const float* __restrict__ b2, float* __restrict__ out, int n)
{
    int wid  = (blockIdx.x * blockDim.x + threadIdx.x) >> 5;
    int lane = threadIdx.x & 31;
    if (wid >= n) return;
    float xv = h[(size_t)wid * CH + lane];
    float dot = 0.f;
#pragma unroll
    for (int j = 0; j < 4; ++j) {
        int o = lane + j * 32;      // 128 hidden channels, 4 per lane
        float a = b1[o];
#pragma unroll
        for (int i = 0; i < CH; ++i)
            a += __shfl(xv, i, 32) * W1[i * 128 + o];
        a = fmaxf(a, 0.f);
        dot += a * W2[o];
    }
#pragma unroll
    for (int off = 16; off > 0; off >>= 1) dot += __shfl_down(dot, off, 32);
    if (lane == 0) out[wid] = dot + b2[0];
}

// ============================================================================
extern "C" void kernel_launch(void* const* d_in, const int* in_sizes, int n_in,
                              void* d_out, int out_size, void* d_ws, size_t ws_size,
                              hipStream_t stream)
{
    (void)in_sizes; (void)n_in; (void)out_size; (void)ws_size;

    const int NTOT = 21504;
    const int ED0 = 32768, ED1 = 8192;               // down
    const int EM0 = 65536, EM1 = 32768, EM2 = 8192;  // mid
    const int EU0 = 32768, EU1 = 8192;               // up
    const int EDT = 40960, EMT = 106496, EUT = 40960;

    auto F = [&](int i) { return (const float*)d_in[i]; };
    const float* x       = F(0);
    const float* ea_down = F(1);
    const float* ea_mid  = F(2);
    const float* ea_up   = F(3);
    const int* ei_down = (const int*)d_in[47];
    const int* ei_mid  = (const int*)d_in[48];
    const int* ei_up   = (const int*)d_in[49];

    // ---- workspace carve (needs ~408 MB) ----
    char* ws = (char*)d_ws;
    size_t off = 0;
    auto carve = [&](size_t bytes) -> char* {
        char* p = ws + off; off += (bytes + 255) & ~(size_t)255; return p;
    };
    float*    h   = (float*)carve((size_t)NTOT * CH * 4);
    float*    s   = (float*)carve((size_t)NTOT * CH * 4);
    float*    cnt = (float*)carve((size_t)7 * NTOT * 4);
    _Float16* z1  = (_Float16*)carve((size_t)65536 * 128 * 2);
    _Float16* z2  = (_Float16*)carve((size_t)65536 * 128 * 2);
    _Float16* wt_d0  = (_Float16*)carve((size_t)1024 * 64 * 2);
    _Float16* wt_d1  = (_Float16*)carve((size_t)1024 * 32 * 2);
    _Float16* wt_m0a = (_Float16*)carve((size_t)128 * 128 * 2);
    _Float16* wt_m0b = (_Float16*)carve((size_t)1024 * 128 * 2);
    _Float16* wt_m1a = (_Float16*)carve((size_t)64 * 64 * 2);
    _Float16* wt_m1b = (_Float16*)carve((size_t)1024 * 64 * 2);
    _Float16* wt_m2a = (_Float16*)carve((size_t)32 * 32 * 2);
    _Float16* wt_m2b = (_Float16*)carve((size_t)1024 * 32 * 2);
    _Float16* wt_u0  = (_Float16*)carve((size_t)1024 * 64 * 2);
    _Float16* wt_u1  = (_Float16*)carve((size_t)1024 * 32 * 2);
    _Float16* we_d0 = (_Float16*)carve((size_t)ED0 * 1024 * 2);
    _Float16* we_d1 = (_Float16*)carve((size_t)ED1 * 1024 * 2);
    _Float16* we_m0 = (_Float16*)carve((size_t)EM0 * 1024 * 2);
    _Float16* we_m1 = (_Float16*)carve((size_t)EM1 * 1024 * 2);
    _Float16* we_m2 = (_Float16*)carve((size_t)EM2 * 1024 * 2);
    _Float16* we_u0 = (_Float16*)carve((size_t)EU0 * 1024 * 2);
    _Float16* we_u1 = (_Float16*)carve((size_t)EU1 * 1024 * 2);

    auto cdiv = [](long a, long b) { return (int)((a + b - 1) / b); };

    auto PREP = [&](const float* W, _Float16* Wt, int K, int N) {
        k_w2h_t<<<cdiv((long)K * N, 256), 256, 0, stream>>>(W, Wt, K, N);
    };
    auto L1 = [&](const float* attr, const float* W, const float* b,
                  _Float16* Y, int E, int H) {
        k_mlp_l1<<<cdiv((long)E * H, 256), 256, 0, stream>>>(attr, W, b, Y, E, H);
    };
    // N=32 -> 2 tiles/wave (ntiles=2); otherwise 4 tiles/wave (ntiles%4==0).
    auto GEMM = [&](const _Float16* X, const _Float16* Wt, const float* bias,
                    _Float16* Y, int M, int K, int N, int relu) {
        int mtiles = M / 16, ntiles = N / 16;
        int nt = (ntiles % 4 == 0) ? 4 : 2;
        int ngroups = ntiles / nt;
        long waves = (long)mtiles * ngroups;
        int blocks = cdiv(waves * 32, 256);
        if (nt == 4) {
            if (relu) k_gemm_wmma<4, true ><<<blocks, 256, 0, stream>>>(X, Wt, bias, Y, M, K, N, ngroups);
            else      k_gemm_wmma<4, false><<<blocks, 256, 0, stream>>>(X, Wt, bias, Y, M, K, N, ngroups);
        } else {
            if (relu) k_gemm_wmma<2, true ><<<blocks, 256, 0, stream>>>(X, Wt, bias, Y, M, K, N, ngroups);
            else      k_gemm_wmma<2, false><<<blocks, 256, 0, stream>>>(X, Wt, bias, Y, M, K, N, ngroups);
        }
    };
    auto CONV = [&](const int* sp, const int* tp, const _Float16* We, int E) {
        (void)hipMemsetAsync(s, 0, (size_t)NTOT * CH * 4, stream);
        k_conv<<<cdiv((long)E * 32, 256), 256, 0, stream>>>(h, sp, tp, We, s, E);
    };
    auto COMB = [&](const float* c) {
        k_combine_relu<<<cdiv((long)NTOT * CH, 256), 256, 0, stream>>>(h, s, c, NTOT * CH);
    };
    auto MID = [&](const float* c, const float* root, int p0, int n) {
        k_combine_mid<<<cdiv((long)n * 32, 256), 256, 0, stream>>>(h, s, c, root, p0, n);
    };

    // ---- 1) transpose+halve all WMMA weights -------------------------------
    PREP(F(8),  wt_d0, 64, 1024);   PREP(F(12), wt_d1, 32, 1024);
    PREP(F(16), wt_m0a, 128, 128);  PREP(F(18), wt_m0b, 128, 1024);
    PREP(F(22), wt_m1a, 64, 64);    PREP(F(24), wt_m1b, 64, 1024);
    PREP(F(28), wt_m2a, 32, 32);    PREP(F(30), wt_m2b, 32, 1024);
    PREP(F(37), wt_u0, 64, 1024);   PREP(F(41), wt_u1, 32, 1024);

    // ---- 2) per-edge kernel MLPs -> We (f16), computed once ----------------
    L1(ea_down,                 F(6),  F(7),  z1, ED0, 64);
    GEMM(z1, wt_d0, F(9),  we_d0, ED0, 64, 1024, 0);
    L1(ea_down + (size_t)ED0*6, F(10), F(11), z1, ED1, 32);
    GEMM(z1, wt_d1, F(13), we_d1, ED1, 32, 1024, 0);

    L1(ea_mid,                  F(14), F(15), z1, EM0, 128);
    GEMM(z1, wt_m0a, F(17), z2, EM0, 128, 128, 1);
    GEMM(z2, wt_m0b, F(19), we_m0, EM0, 128, 1024, 0);
    L1(ea_mid + (size_t)EM0*6,  F(20), F(21), z1, EM1, 64);
    GEMM(z1, wt_m1a, F(23), z2, EM1, 64, 64, 1);
    GEMM(z2, wt_m1b, F(25), we_m1, EM1, 64, 1024, 0);
    L1(ea_mid + (size_t)(EM0+EM1)*6, F(26), F(27), z1, EM2, 32);
    GEMM(z1, wt_m2a, F(29), z2, EM2, 32, 32, 1);
    GEMM(z2, wt_m2b, F(31), we_m2, EM2, 32, 1024, 0);

    L1(ea_up,                   F(35), F(36), z1, EU0, 64);
    GEMM(z1, wt_u0, F(38), we_u0, EU0, 64, 1024, 0);
    L1(ea_up + (size_t)EU0*6,   F(39), F(40), z1, EU1, 32);
    GEMM(z1, wt_u1, F(42), we_u1, EU1, 32, 1024, 0);

    // ---- 3) scatter degrees (fixed over depth) -----------------------------
    const int *sd0 = ei_down,        *td0 = ei_down + EDT;
    const int *sd1 = ei_down + ED0,  *td1 = ei_down + EDT + ED0;
    const int *sm0 = ei_mid,         *tm0 = ei_mid + EMT;
    const int *sm1 = ei_mid + EM0,   *tm1 = ei_mid + EMT + EM0;
    const int *sm2 = ei_mid + EM0 + EM1, *tm2 = ei_mid + EMT + EM0 + EM1;
    const int *su0 = ei_up,          *tu0 = ei_up + EUT;
    const int *su1 = ei_up + EU0,    *tu1 = ei_up + EUT + EU0;
    float *c_d0 = cnt,            *c_d1 = cnt + NTOT,   *c_u0 = cnt + 2*NTOT,
          *c_u1 = cnt + 3*NTOT,   *c_m0 = cnt + 4*NTOT, *c_m1 = cnt + 5*NTOT,
          *c_m2 = cnt + 6*NTOT;
    (void)hipMemsetAsync(cnt, 0, (size_t)7 * NTOT * 4, stream);
    k_count<<<cdiv(ED0, 256), 256, 0, stream>>>(td0, c_d0, ED0);
    k_count<<<cdiv(ED1, 256), 256, 0, stream>>>(td1, c_d1, ED1);
    k_count<<<cdiv(EU0, 256), 256, 0, stream>>>(tu0, c_u0, EU0);
    k_count<<<cdiv(EU1, 256), 256, 0, stream>>>(tu1, c_u1, EU1);
    k_count<<<cdiv(EM0, 256), 256, 0, stream>>>(tm0, c_m0, EM0);
    k_count<<<cdiv(EM1, 256), 256, 0, stream>>>(tm1, c_m1, EM1);
    k_count<<<cdiv(EM2, 256), 256, 0, stream>>>(tm2, c_m2, EM2);

    // ---- 4) fc_in ----------------------------------------------------------
    k_fc_in<<<cdiv((long)NTOT * CH, 256), 256, 0, stream>>>(x, F(4), F(5), h, NTOT);

    // ---- 5) DEPTH=2 V-cycles ----------------------------------------------
    for (int d = 0; d < 2; ++d) {
        CONV(sd0, td0, we_d0, ED0);  COMB(c_d0);                       // down l=0
        CONV(sd1, td1, we_d1, ED1);  COMB(c_d1);                       // down l=1
        CONV(sm2, tm2, we_m2, EM2);  MID(c_m2, F(34), 20480, 1024);    // mid  l=2
        CONV(su1, tu1, we_u1, EU1);  COMB(c_u1);                       // up   l=1
        CONV(sm1, tm1, we_m1, EM1);  MID(c_m1, F(33), 16384, 4096);    // mid  l=1
        CONV(su0, tu0, we_u0, EU0);  COMB(c_u0);                       // up   l=0
        CONV(sm0, tm0, we_m0, EM0);  MID(c_m0, F(32), 0, 16384);       // mid  l=0
    }

    // ---- 6) output head ----------------------------------------------------
    k_fc_out<<<cdiv((long)16384 * 32, 256), 256, 0, stream>>>(
        h, F(43), F(44), F(45), F(46), (float*)d_out, 16384);
}
